// DeepMG_28681791603121
// MI455X (gfx1250) — matrix operand order repaired and checked by reading, not compiled, not run on hardware
//
#include <hip/hip_runtime.h>

// ---------------------------------------------------------------------------
// Types for CDNA5 WMMA (wave32): bf16 16x16x32, f32 accumulate
// ---------------------------------------------------------------------------
typedef __attribute__((ext_vector_type(16))) __bf16 v16bf;
typedef __attribute__((ext_vector_type(8)))  __bf16 v8bf;
typedef __attribute__((ext_vector_type(8)))  float  v8f;

#define BM 128
#define BN 128
#define KT 32

// Tensor Data Mover availability (arity differs across toolchains)
#if defined(__has_builtin)
#if __has_builtin(__builtin_amdgcn_tensor_load_to_lds)
#define USE_TDM 1
#endif
#endif

// ---------------------------------------------------------------------------
// f32 -> bf16 (round to nearest even)
// ---------------------------------------------------------------------------
__device__ __forceinline__ unsigned short f32_to_bf16(float f) {
  unsigned int u = __float_as_uint(f);
  unsigned int r = u + 0x7FFFu + ((u >> 16) & 1u);
  return (unsigned short)(r >> 16);
}

__global__ void k_f32_to_bf16(const float* __restrict__ s,
                              unsigned short* __restrict__ d, long n) {
  long i = (long)blockIdx.x * blockDim.x + threadIdx.x;
  if (i < n) d[i] = f32_to_bf16(s[i]);
}

// W[K,N] f32 -> Wt[N,K] bf16, LDS-tiled transpose (coalesced both sides)
__global__ void k_f32_to_bf16_T(const float* __restrict__ W,
                                unsigned short* __restrict__ Wt, int K, int N) {
  __shared__ unsigned short tile[32][33];
  int kb = blockIdx.y * 32, nb = blockIdx.x * 32;
  int tx = threadIdx.x & 31, ty = threadIdx.x >> 5;   // 32 x 8
#pragma unroll
  for (int i = ty; i < 32; i += 8) {
    int k = kb + i, n = nb + tx;
    float v = (k < K && n < N) ? W[(size_t)k * N + n] : 0.0f;
    tile[i][tx] = f32_to_bf16(v);
  }
  __syncthreads();
#pragma unroll
  for (int i = ty; i < 32; i += 8) {
    int n = nb + i, k = kb + tx;
    if (n < N && k < K) Wt[(size_t)n * K + k] = tile[tx][i];
  }
}

__global__ void k_fill(float* __restrict__ p, float v, long n) {
  long i = (long)blockIdx.x * blockDim.x + threadIdx.x;
  if (i < n) p[i] = v;
}

// ---------------------------------------------------------------------------
// TDM helper: issue a 2D tensor tile load global->LDS (bf16 elements).
// D# group layout per CDNA5 ISA 8.3/8.4 (2D: groups 2/3 NULL/zero).
// ---------------------------------------------------------------------------
#ifdef USE_TDM
typedef __attribute__((ext_vector_type(4))) unsigned int u32x4;
typedef __attribute__((ext_vector_type(4))) int i32x4;
typedef __attribute__((ext_vector_type(8))) int i32x8;

__device__ __forceinline__ void tdm_load_tile_bf16(
    unsigned lds_off, const void* gaddr,
    unsigned tdim0, unsigned tdim1,          // remaining tensor extent (elems)
    unsigned tile0, unsigned tile1,          // tile dims (elems)
    unsigned long long stride0) {            // row stride (elems)
  unsigned long long ga = (unsigned long long)gaddr;
  u32x4 g0;
  g0[0] = 1u;                                        // count=1, user descriptor
  g0[1] = lds_off;                                   // lds_addr
  g0[2] = (unsigned)ga;                              // global_addr[31:0]
  g0[3] = (unsigned)((ga >> 32) & 0x01FFFFFFu) | (2u << 30);  // addr hi, type=2
  i32x8 g1;
  g1[0] = (int)(1u << 16);                           // data_size=1 (2 bytes)
  g1[1] = (int)((tdim0 & 0xFFFFu) << 16);            // tensor_dim0[15:0]
  g1[2] = (int)((tdim0 >> 16) | ((tdim1 & 0xFFFFu) << 16));
  g1[3] = (int)((tdim1 >> 16) | (tile0 << 16));      // tile_dim0
  g1[4] = (int)(tile1 & 0xFFFFu);                    // tile_dim1 (tile_dim2=0)
  g1[5] = (int)(unsigned)(stride0 & 0xFFFFFFFFu);    // tensor_dim0_stride lo
  g1[6] = (int)(unsigned)((stride0 >> 32) & 0xFFFFu);
  g1[7] = 0;
  i32x4 gz = {0, 0, 0, 0};
#if __clang_major__ >= 23
  i32x8 gz8 = {0, 0, 0, 0, 0, 0, 0, 0};
  __builtin_amdgcn_tensor_load_to_lds(g0, g1, gz, gz, gz8, 0);
#else
  __builtin_amdgcn_tensor_load_to_lds(g0, g1, gz, gz, 0);
#endif
}
#endif

// ---------------------------------------------------------------------------
// bf16 WMMA GEMM:  C[M,N] = A[M,K] @ Wt[N,K]^T + bias[N]   (A,Wt bf16, C f32)
// Block tile 128x128, 8 waves (4x2), each wave 32x64 (2x4 wmma tiles).
// TDM path: double-buffered LDS; wave 0 streams tile t+1 via the Tensor Data
// Mover while all waves compute on tile t (TENSORcnt<=2 => current tile done,
// since per-wave TDM ops complete in order).
// Requires: K % 32 == 0, N % 128 == 0. M guarded (TDM: OOB reads return 0).
// ---------------------------------------------------------------------------
__launch_bounds__(256)
__global__ void k_gemm_bf16_wmma(const unsigned short* __restrict__ Ax,
                                 const unsigned short* __restrict__ Btx,
                                 const float* __restrict__ bias,
                                 float* __restrict__ C,
                                 int M, int N, int K) {
  const __bf16* __restrict__ A  = (const __bf16*)Ax;
  const __bf16* __restrict__ Bt = (const __bf16*)Btx;   // [N][K]

  __shared__ __bf16 sA[2][BM * KT];   // [128][32] row-major (rows = M)
  __shared__ __bf16 sB[2][BN * KT];   // [128][32] row-major (rows = N)

  const int tid  = threadIdx.x;
  const int lane = tid & 31;
  const int wave = tid >> 5;
  const int wm   = wave >> 1;      // 0..3 -> 32-row strip
  const int wn   = wave & 1;       // 0..1 -> 64-col strip
  const int m0   = blockIdx.y * BM;
  const int n0   = blockIdx.x * BN;

  const int fr = lane & 15;        // fragment row (A) / column (B)
  const int kh = lane >> 4;        // K-half select (contiguous k order used by this code)

  v8f acc[2][4] = {};

  const int nk = K / KT;

#ifdef USE_TDM
  if (wave == 0) {
    tdm_load_tile_bf16((unsigned)(unsigned long long)&sA[0][0],
                       A + (size_t)m0 * K, (unsigned)K, (unsigned)(M - m0),
                       KT, BM, (unsigned long long)K);
    tdm_load_tile_bf16((unsigned)(unsigned long long)&sB[0][0],
                       Bt + (size_t)n0 * K, (unsigned)K, (unsigned)(N - n0),
                       KT, BN, (unsigned long long)K);
  }
#endif

  for (int t = 0; t < nk; ++t) {
    const int buf = t & 1;
    const int k0  = t * KT;

#ifdef USE_TDM
    if (wave == 0) {
      if (t + 1 < nk) {
        // stream next tile into the other buffer while we compute on this one
        int k1 = k0 + KT;
        tdm_load_tile_bf16((unsigned)(unsigned long long)&sA[buf ^ 1][0],
                           A + (size_t)m0 * K + k1,
                           (unsigned)(K - k1), (unsigned)(M - m0),
                           KT, BM, (unsigned long long)K);
        tdm_load_tile_bf16((unsigned)(unsigned long long)&sB[buf ^ 1][0],
                           Bt + (size_t)n0 * K + k1,
                           (unsigned)(K - k1), (unsigned)(N - n0),
                           KT, BN, (unsigned long long)K);
        __builtin_amdgcn_s_wait_tensorcnt(2);   // current tile's pair landed
      } else {
        __builtin_amdgcn_s_wait_tensorcnt(0);
      }
    }
    __syncthreads();
#else
    // --- manual staging: pure b128 copies, no scalar scatter ---
#pragma unroll
    for (int it = 0; it < 2; ++it) {
      int idx = (tid + it * 256) * 8;      // bf16 element index in tile
      int r = idx >> 5;
      int c = idx & 31;
      int gr = m0 + r;
      uint4 v = make_uint4(0u, 0u, 0u, 0u);
      if (gr < M) v = *(const uint4*)(A + (size_t)gr * K + k0 + c);
      *(uint4*)(&sA[0][r * KT + c]) = v;
    }
#pragma unroll
    for (int it = 0; it < 2; ++it) {
      int idx = (tid + it * 256) * 8;
      int r = idx >> 5;                    // n within tile (N % 128 == 0: in range)
      int c = idx & 31;
      uint4 v = *(const uint4*)(Bt + (size_t)(n0 + r) * K + k0 + c);
      *(uint4*)(&sB[0][r * KT + c]) = v;
    }
    __syncthreads();
    if (k0 + KT < K) {
      __builtin_prefetch(A + (size_t)(m0 + (tid >> 1)) * K + (k0 + KT), 0, 0);
      __builtin_prefetch(Bt + (size_t)(n0 + (tid >> 1)) * K + (k0 + KT), 0, 0);
    }
#endif

    const __bf16* __restrict__ pA = &sA[buf][0];
    const __bf16* __restrict__ pB = &sB[buf][0];
#ifndef USE_TDM
    pA = &sA[0][0];
    pB = &sB[0][0];
#endif

    // --- build fragments (contiguous k = 16*kh + s, same order for A and B) ---
    v16bf af[2], bfv[4];
#pragma unroll
    for (int mi = 0; mi < 2; ++mi) {
      const __bf16* base = &pA[(wm * 32 + mi * 16 + fr) * KT + kh * 16];
      v8bf lo = *(const v8bf*)base;         // K = 16*kh .. +7
      v8bf hi = *(const v8bf*)(base + 8);   // K = 16*kh + 8 .. +7
      v16bf a;
#pragma unroll
      for (int j = 0; j < 8; ++j) { a[j] = lo[j]; a[8 + j] = hi[j]; }
      af[mi] = a;
    }
#pragma unroll
    for (int ni = 0; ni < 4; ++ni) {
      // lane<16: column fr, K 0..15 ; lane>=16: column fr, K 16..31
      bfv[ni] = *(const v16bf*)&pB[(wn * 64 + ni * 16 + fr) * KT + kh * 16];
    }

#pragma unroll
    for (int mi = 0; mi < 2; ++mi)
#pragma unroll
      for (int ni = 0; ni < 4; ++ni)
        acc[mi][ni] = __builtin_amdgcn_wmma_f32_16x16x32_bf16(
            false, af[mi], false, bfv[ni], (short)0, acc[mi][ni], false, false);

    __syncthreads();   // all reads of `buf` done before it is re-streamed
  }

  // --- epilogue: bias add + store (C-fragment: vgpr r -> row r + 8*kh) ---
#pragma unroll
  for (int mi = 0; mi < 2; ++mi) {
#pragma unroll
    for (int ni = 0; ni < 4; ++ni) {
      int col = n0 + wn * 64 + ni * 16 + fr;
      float bv = bias ? bias[col] : 0.0f;
#pragma unroll
      for (int r = 0; r < 8; ++r) {
        int rowg = m0 + wm * 32 + mi * 16 + kh * 8 + r;
        if (rowg < M) C[(size_t)rowg * N + col] = acc[mi][ni][r] + bv;
      }
    }
  }
}

// ---------------------------------------------------------------------------
// GATv2 edge phase
// ---------------------------------------------------------------------------
__device__ __forceinline__ void atomicMaxF(float* addr, float v) {
  unsigned int* a = (unsigned int*)addr;
  unsigned int old = *a;
  while (__uint_as_float(old) < v) {
    unsigned int assumed = old;
    old = atomicCAS(a, assumed, __float_as_uint(v));
    if (old == assumed) break;
  }
}

// alpha[e,h] = sum_c att[h,c] * lrelu(xl[src] + xr[dst] + (ea @ We))
__global__ void k_edge_alpha(const float* __restrict__ xl, const float* __restrict__ xr,
                             const float* __restrict__ ea, const float* __restrict__ We,
                             const float* __restrict__ att, const int* __restrict__ ei,
                             float* __restrict__ alpha, int E, int H, int C, int edim) {
  int e = blockIdx.x;
  int HC = H * C;
  __shared__ float sacc[8];
  if (threadIdx.x < H) sacc[threadIdx.x] = 0.0f;
  __syncthreads();
  int src = ei[e], dst = ei[E + e];
  int h  = threadIdx.x >> 5;   // 32 lanes per head (H==8, block 256)
  int c0 = threadIdx.x & 31;
  float part = 0.0f;
  for (int c = c0; c < C; c += 32) {
    int hc = h * C + c;
    float ev;
    if (edim == 1) {
      ev = ea[e] * We[hc];                     // outer product, never materialized
    } else {
      ev = 0.0f;
      for (int j = 0; j < edim; ++j) ev += ea[(size_t)e * edim + j] * We[(size_t)j * HC + hc];
    }
    float m = xl[(size_t)src * HC + hc] + xr[(size_t)dst * HC + hc] + ev;
    m = m > 0.0f ? m : 0.2f * m;
    part += att[hc] * m;
  }
  atomicAdd(&sacc[h], part);                   // ds_add_f32
  __syncthreads();
  if (threadIdx.x < H) alpha[(size_t)e * H + threadIdx.x] = sacc[threadIdx.x];
}

__global__ void k_edge_amax(const float* __restrict__ alpha, const int* __restrict__ ei,
                            float* __restrict__ amax, int E, int H) {
  long i = (long)blockIdx.x * blockDim.x + threadIdx.x;
  if (i >= (long)E * H) return;
  int e = (int)(i / H), h = (int)(i % H);
  int dst = ei[E + e];
  atomicMaxF(&amax[(size_t)dst * H + h], alpha[i]);
}

__global__ void k_edge_expden(float* __restrict__ alpha, const int* __restrict__ ei,
                              const float* __restrict__ amax, float* __restrict__ den,
                              int E, int H) {
  long i = (long)blockIdx.x * blockDim.x + threadIdx.x;
  if (i >= (long)E * H) return;
  int e = (int)(i / H), h = (int)(i % H);
  int dst = ei[E + e];
  float ex = __expf(alpha[i] - amax[(size_t)dst * H + h]);
  alpha[i] = ex;
  atomicAdd(&den[(size_t)dst * H + h], ex);
}

__global__ void k_edge_scatter(const float* __restrict__ xl, const float* __restrict__ ex,
                               const float* __restrict__ den, const int* __restrict__ ei,
                               float* __restrict__ out, int E, int H, int C) {
  int e = blockIdx.x;
  int HC = H * C;
  __shared__ float sw[8];
  int src = ei[e], dst = ei[E + e];
  if (threadIdx.x < H)
    sw[threadIdx.x] = ex[(size_t)e * H + threadIdx.x] /
                      (den[(size_t)dst * H + threadIdx.x] + 1e-16f);
  __syncthreads();
  for (int hc = threadIdx.x; hc < HC; hc += blockDim.x)
    atomicAdd(&out[(size_t)dst * HC + hc], sw[hc / C] * xl[(size_t)src * HC + hc]);
}

__global__ void k_bias_add(float* __restrict__ x, const float* __restrict__ bias,
                           int Nn, int HC) {
  long i = (long)blockIdx.x * blockDim.x + threadIdx.x;
  if (i < (long)Nn * HC) x[i] += bias[i % HC];
}

// mean pool into the combined feature buffer at column offset
__global__ void k_pool(const float* __restrict__ x, const int* __restrict__ batch,
                       float* __restrict__ comb, int Nn, int HC, int colOff,
                       float invCnt) {
  long i = (long)blockIdx.x * blockDim.x + threadIdx.x;
  if (i >= (long)Nn * HC) return;
  int n = (int)(i / HC), c = (int)(i % HC);
  int b = batch[n];
  atomicAdd(&comb[(size_t)b * 5632 + colOff + c], x[i] * invCnt);
}

// ---------------------------------------------------------------------------
// MLP head (M=8, negligible FLOPs): one thread per output element
// mode 0: BN(eval) + ReLU, mode 1: sigmoid
// ---------------------------------------------------------------------------
__global__ void k_mlp(const float* __restrict__ in, const float* __restrict__ W,
                      const float* __restrict__ bias, const float* __restrict__ g,
                      const float* __restrict__ beta, const float* __restrict__ mean,
                      const float* __restrict__ var, float* __restrict__ out,
                      int M, int K, int N, int mode) {
  long i = (long)blockIdx.x * blockDim.x + threadIdx.x;
  if (i >= (long)M * N) return;
  int m = (int)(i / N), n = (int)(i % N);
  float acc = bias[n];
  const float* row = in + (size_t)m * K;
  for (int k = 0; k < K; ++k) acc += row[k] * W[(size_t)k * N + n];
  if (mode == 0) {
    acc = (acc - mean[n]) * rsqrtf(var[n] + 1e-5f) * g[n] + beta[n];
    acc = acc > 0.0f ? acc : 0.0f;
  } else {
    acc = 1.0f / (1.0f + __expf(-acc));
  }
  out[i] = acc;
}

// ---------------------------------------------------------------------------
// Host orchestration
// ---------------------------------------------------------------------------
extern "C" void kernel_launch(void* const* d_in, const int* in_sizes, int n_in,
                              void* d_out, int out_size, void* d_ws, size_t ws_size,
                              hipStream_t stream) {
  (void)n_in; (void)out_size; (void)ws_size;

  char* ws = (char*)d_ws;
  size_t off = 0;
  auto carve = [&](size_t bytes) -> char* {
    char* p = ws + off;
    off = (off + bytes + 255) & ~(size_t)255;
    return p;
  };

  unsigned short* bfX = (unsigned short*)carve((size_t)4000 * 2560 * 2);
  unsigned short* bfW = (unsigned short*)carve((size_t)2560 * 2560 * 2);  // Wt [N][K]
  float* xl    = (float*)carve((size_t)4000 * 2560 * 4);
  float* xr    = (float*)carve((size_t)4000 * 2560 * 4);
  float* h1    = (float*)carve((size_t)4000 * 2560 * 4);
  float* h2    = (float*)carve((size_t)4000 * 2560 * 4);
  float* alpha = (float*)carve((size_t)32000 * 8 * 4);
  float* amax  = (float*)carve((size_t)4000 * 8 * 4);
  float* den   = (float*)carve((size_t)4000 * 8 * 4);
  float* comb  = (float*)carve((size_t)8 * 5632 * 4);
  float* m1    = (float*)carve((size_t)8 * 2560 * 4);
  float* m2    = (float*)carve((size_t)8 * 1280 * 4);
  float* m3    = (float*)carve((size_t)8 * 640 * 4);
  float* m4    = (float*)carve((size_t)8 * 320 * 4);

  auto F = [&](int i) { return (const float*)d_in[i]; };
  auto I = [&](int i) { return (const int*)d_in[i]; };
  auto cdiv = [](long a, long b) { return (int)((a + b - 1) / b); };

  auto conv = [&](const float* s, unsigned short* d, long n) {
    k_f32_to_bf16<<<cdiv(n, 256), 256, 0, stream>>>(s, d, n);
  };
  auto convT = [&](const float* W, unsigned short* Wt, int K, int N) {
    dim3 grid(cdiv(N, 32), cdiv(K, 32));
    k_f32_to_bf16_T<<<grid, 256, 0, stream>>>(W, Wt, K, N);
  };
  auto fill = [&](float* p, float v, long n) {
    k_fill<<<cdiv(n, 256), 256, 0, stream>>>(p, v, n);
  };
  auto gemm = [&](const unsigned short* Abf, const unsigned short* Wt,
                  const float* bias, float* Cc, int M, int N, int K) {
    dim3 grid(N / BN, cdiv(M, BM));
    k_gemm_bf16_wmma<<<grid, 256, 0, stream>>>(Abf, Wt, bias, Cc, M, N, K);
  };

  // one GATv2 layer (eval mode): Wl/bl/Wr/br/We/att/bias at params[pbase..pbase+6]
  auto gat_layer = [&](const float* xin, int Nn, int Fin, int H, int C,
                       const int* ei, int E, const float* ea, int edim,
                       int pbase, float* outbuf) {
    int HC = H * C;
    conv(xin, bfX, (long)Nn * Fin);
    convT(F(pbase + 0), bfW, Fin, HC);
    gemm(bfX, bfW, F(pbase + 1), xl, Nn, HC, Fin);     // xl = x@Wl + bl
    convT(F(pbase + 2), bfW, Fin, HC);
    gemm(bfX, bfW, F(pbase + 3), xr, Nn, HC, Fin);     // xr = x@Wr + br
    fill(outbuf, 0.0f, (long)Nn * HC);
    fill(amax, -1e30f, (long)Nn * H);
    fill(den, 0.0f, (long)Nn * H);
    k_edge_alpha<<<E, 256, 0, stream>>>(xl, xr, ea, F(pbase + 4), F(pbase + 5),
                                        ei, alpha, E, H, C, edim);
    k_edge_amax<<<cdiv((long)E * H, 256), 256, 0, stream>>>(alpha, ei, amax, E, H);
    k_edge_expden<<<cdiv((long)E * H, 256), 256, 0, stream>>>(alpha, ei, amax, den, E, H);
    k_edge_scatter<<<E, 256, 0, stream>>>(xl, alpha, den, ei, outbuf, E, H, C);
    k_bias_add<<<cdiv((long)Nn * HC, 256), 256, 0, stream>>>(outbuf, F(pbase + 6), Nn, HC);
  };

  auto branch = [&](int xi, int eii, int eai, int bi, int Nn, int Fin, int H, int C,
                    int edim, int pa, int pb, int colOff, float invCnt) {
    int E = in_sizes[eii] / 2;
    gat_layer(F(xi), Nn, Fin, H, C, I(eii), E, F(eai), edim, pa, h1);
    gat_layer(h1, Nn, H * C, H, C, I(eii), E, F(eai), edim, pb, h2);
    k_pool<<<cdiv((long)Nn * H * C, 256), 256, 0, stream>>>(h2, I(bi), comb, Nn,
                                                            H * C, colOff, invCnt);
  };

  fill(comb, 0.0f, 8L * 5632);

  // params layout (setup_inputs insertion order): 12 data arrays, then
  // d1@12, d2@19, t0a@26, t0b@33, t1a@40, t1b@47, mlp Ws@54 bs@59 gs@64 betas@68 ms@72 vs@76
  branch(0, 1, 2, 3,   384,  512, 8,  64, 7, 12, 19, 0,          1.0f / 48.0f);
  branch(4, 5, 6, 7,   4000, 2560, 8, 320, 1, 26, 33, 512,        1.0f / 500.0f);
  branch(8, 9, 10, 11, 4000, 2560, 8, 320, 1, 40, 47, 512 + 2560, 1.0f / 500.0f);

  // MLP head: 5632 -> 2560 -> 1280 -> 640 -> 320 -> 1
  const int dims[6] = {5632, 2560, 1280, 640, 320, 1};
  float* bufs[6] = {comb, m1, m2, m3, m4, (float*)d_out};
  for (int i = 0; i < 5; ++i) {
    int Kd = dims[i], Nd = dims[i + 1];
    int mode = (i < 4) ? 0 : 1;
    const float* g  = (i < 4) ? F(64 + i) : nullptr;
    const float* be = (i < 4) ? F(68 + i) : nullptr;
    const float* mn = (i < 4) ? F(72 + i) : nullptr;
    const float* vr = (i < 4) ? F(76 + i) : nullptr;
    k_mlp<<<cdiv(8L * Nd, 256), 256, 0, stream>>>(bufs[i], F(54 + i), F(59 + i),
                                                  g, be, mn, vr, bufs[i + 1],
                                                  8, Kd, Nd, mode);
  }
}
